// DualMemoryElmanCellOptimized_2147483648050
// MI455X (gfx1250) — matrix operand order repaired and checked by reading, not compiled, not run on hardware
//
#include <hip/hip_runtime.h>
#include <cstdint>
#include <cstddef>

#define Bsz 32
#define Tsz 1024
#define Dsz 512
#define Nsz 64

typedef __attribute__((ext_vector_type(16))) __bf16 v16bf;
typedef __attribute__((ext_vector_type(8)))  __bf16 v8bf;
typedef __attribute__((ext_vector_type(8)))  float  v8f;

// ---------------------------------------------------------------------------
// WMMA helpers (CDNA5: V_WMMA_F32_16X16X32_BF16, wave32)
// ---------------------------------------------------------------------------
__device__ __forceinline__ v8f wmma_bf16_f32(v16bf a, v16bf b, v8f c) {
  // 8 args: (neg_a, A, neg_b, B, c_mod, C, reuse_a, reuse_b)
  return __builtin_amdgcn_wmma_f32_16x16x32_bf16(false, a, false, b,
                                                 (short)0, c, false, false);
}

// A-matrix 16x32 bf16 per-lane layout (ISA 7.12.2):
//   lane = hf*16 + m ; element e -> K = e + 8*hf + (e>=8 ? 8 : 0)
// i.e. two contiguous 8-element chunks: [k0+8*hf, +8) and [k0+16+8*hf, +8)
__device__ __forceinline__ v16bf load_a_lds(const __bf16* rowp, int k0, int hf) {
  const v8bf lo = *(const v8bf*)(rowp + k0 + 8 * hf);
  const v8bf hi = *(const v8bf*)(rowp + k0 + 16 + 8 * hf);
  v16bf a;
#pragma unroll
  for (int e = 0; e < 8; ++e) { a[e] = lo[e]; a[8 + e] = hi[e]; }
  return a;
}

// Same A layout but sourced from f32 global memory (converted on the fly).
__device__ __forceinline__ v16bf load_a_f32(const float* rowp, int k0, int hf) {
  v16bf a;
#pragma unroll
  for (int e = 0; e < 8; ++e) a[e] = (__bf16)rowp[k0 + 8 * hf + e];
#pragma unroll
  for (int e = 0; e < 8; ++e) a[8 + e] = (__bf16)rowp[k0 + 16 + 8 * hf + e];
  return a;
}

// B-matrix 32x16 bf16: lane = hf*16 + n ; element e -> K = e + 8*hf + (e>=8 ? 8 : 0).
// C[m,n] = sum_k H[m,k] * W[n,k]  (Linear(y)=y@W.T), W row-major [n][k]:
// per-lane B fragment = W row n chunks [k0+8*hf, +8) and [k0+16+8*hf, +8).
__device__ __forceinline__ v16bf load_b_w(const __bf16* W, int n, int k0, int hf) {
  const v8bf lo = *(const v8bf*)(W + (size_t)n * Dsz + k0 + 8 * hf);
  const v8bf hi = *(const v8bf*)(W + (size_t)n * Dsz + k0 + 16 + 8 * hf);
  v16bf b;
#pragma unroll
  for (int e = 0; e < 8; ++e) { b[e] = lo[e]; b[8 + e] = hi[e]; }
  return b;
}

// ---------------------------------------------------------------------------
// gfx1250 async LDS copy (ASYNCcnt path, inline asm for toolchain portability)
// ---------------------------------------------------------------------------
__device__ __forceinline__ void async_g2lds_b128(void* lds_ptr, const void* gptr) {
  // Generic LDS pointers carry the LDS byte offset in addr[31:0] (ISA 10.2).
  const unsigned lds = (unsigned)(uintptr_t)lds_ptr;
  asm volatile("global_load_async_to_lds_b128 %0, %1, off"
               :: "v"(lds), "v"((unsigned long long)(uintptr_t)gptr)
               : "memory");
}

__device__ __forceinline__ void wait_async0() {
  asm volatile("s_wait_asynccnt 0x0" ::: "memory");
}

// ---------------------------------------------------------------------------
// Small utility kernels
// ---------------------------------------------------------------------------
__global__ void f2bf_kernel(const float* __restrict__ s, __bf16* __restrict__ d, int n) {
  int i = blockIdx.x * 256 + threadIdx.x;
  int stride = gridDim.x * 256;
  for (; i < n; i += stride) d[i] = (__bf16)s[i];
}

__global__ void copyf_kernel(const float* __restrict__ s, float* __restrict__ d, long n) {
  long i = (long)blockIdx.x * 256 + threadIdx.x;
  long stride = (long)gridDim.x * 256;
  for (; i < n; i += stride) d[i] = s[i];
}

// ---------------------------------------------------------------------------
// Kernel 1: Wx_all[b*T+t, e] = sum_d x_seq[b,t,d] * W_x[e,d]
// M = B*T = 32768, N = 512, K = 512.
// WG = 256 threads = 8 waves; wave w owns rows [128*blk + 16*w, +16),
// cols [64*blockIdx.y, +64) as 4 WMMA accumulators.
// ---------------------------------------------------------------------------
__global__ void __launch_bounds__(256) wx_gemm_kernel(
    const float* __restrict__ x, const __bf16* __restrict__ Wxbf,
    float* __restrict__ out) {
  const int wave = threadIdx.x >> 5;
  const int lane = threadIdx.x & 31;
  const int hf = lane >> 4;
  const int nloc = lane & 15;
  const long rowBase = (long)blockIdx.x * 128 + wave * 16;
  const int colBase = blockIdx.y * 64;

  const v8f z = {0.f, 0.f, 0.f, 0.f, 0.f, 0.f, 0.f, 0.f};
  v8f acc[4] = {z, z, z, z};

  const float* rowp = x + (rowBase + nloc) * (long)Dsz;
  for (int k0 = 0; k0 < Dsz; k0 += 32) {
    v16bf a = load_a_f32(rowp, k0, hf);
    __builtin_prefetch(rowp + k0 + 32, 0, 3);  // global_prefetch_b8, near
#pragma unroll
    for (int nt = 0; nt < 4; ++nt) {
      v16bf b = load_b_w(Wxbf, colBase + nt * 16 + nloc, k0, hf);
      acc[nt] = wmma_bf16_f32(a, b, acc[nt]);
    }
  }
  // C/D layout: lane holds n = colBase+nt*16+nloc, rows m = i + 8*hf
#pragma unroll
  for (int nt = 0; nt < 4; ++nt) {
    const int n = colBase + nt * 16 + nloc;
#pragma unroll
    for (int i = 0; i < 8; ++i) {
      const long m = rowBase + i + 8 * hf;
      out[m * Dsz + n] = acc[nt][i];
    }
  }
}

// ---------------------------------------------------------------------------
// Recurrent kernel helpers
// ---------------------------------------------------------------------------
// logits[m,n] = scale * dot(h_work[m], tape[b0+m, n, :]); softmax over n.
__device__ __forceinline__ void attn_pass(int tid, int b0,
    const __bf16* s_hw, const float* tape, float* s_logit, float* s_attn) {
  for (int idx = tid; idx < 16 * Nsz; idx += 256) {
    const int m = idx >> 6, n = idx & (Nsz - 1);
    const float4* tr = (const float4*)(tape + ((size_t)(b0 + m) * Nsz + n) * Dsz);
    const __bf16* hw = s_hw + (size_t)m * Dsz;
    float s = 0.f;
#pragma unroll 4
    for (int dq = 0; dq < Dsz / 4; ++dq) {
      const float4 tv = tr[dq];
      s += (float)hw[4 * dq + 0] * tv.x + (float)hw[4 * dq + 1] * tv.y +
           (float)hw[4 * dq + 2] * tv.z + (float)hw[4 * dq + 3] * tv.w;
    }
    s_logit[idx] = s * 0.044194173824159216f;  // 1/sqrt(512)
  }
  __syncthreads();
  if (tid < 16) {
    float mx = -3.0e38f;
    for (int n = 0; n < Nsz; ++n) mx = fmaxf(mx, s_logit[tid * Nsz + n]);
    float sum = 0.f;
    for (int n = 0; n < Nsz; ++n) {
      const float e = __expf(s_logit[tid * Nsz + n] - mx);
      s_attn[tid * Nsz + n] = e;
      sum += e;
    }
    const float inv = 1.f / sum;
    for (int n = 0; n < Nsz; ++n) s_attn[tid * Nsz + n] *= inv;
  }
  __syncthreads();
}

// C[16,512] = s_hw[16,512] @ W.T via 32 N-tiles; wave owns 4 tiles.
__device__ __forceinline__ void gemm16x512(int wave, int lane,
    const __bf16* s_hw, const __bf16* __restrict__ W, v8f acc[4]) {
  const int hf = lane >> 4;
  const int nloc = lane & 15;
  const __bf16* arow = s_hw + (size_t)(lane & 15) * Dsz;
  for (int k0 = 0; k0 < Dsz; k0 += 32) {
    const v16bf a = load_a_lds(arow, k0, hf);
    __builtin_prefetch(W + ((size_t)(wave * 4) * 16 + nloc) * Dsz + k0 + 32, 0, 3);
#pragma unroll
    for (int nt = 0; nt < 4; ++nt) {
      const int n = (wave * 4 + nt) * 16 + nloc;
      const v16bf b = load_b_w(W, n, k0, hf);
      acc[nt] = wmma_bf16_f32(a, b, acc[nt]);
    }
  }
}

// ---------------------------------------------------------------------------
// Kernel 2: the sequential recurrence. 2 workgroups, 16 batch rows each
// (independent chains -> no inter-WG sync). 8 waves, full T loop inside.
// Dynamic LDS partition (88 KB):
//   [0      , 16384) : s_hw     16x512 bf16  (WMMA A source)
//   [16384  , 49152) : s_aux    16x512 f32   (read vec, then write_val)
//   [49152  , 81920) : s_wx     16x512 f32   (async-staged Wx tile for step t)
//   [81920  , 86016) : s_logit  16x64  f32
//   [86016  , 90112) : s_attn   16x64  f32
// ---------------------------------------------------------------------------
#define RECUR_LDS_BYTES 90112

__global__ void __launch_bounds__(256) recurrent_kernel(
    const float* __restrict__ wx_all, const __bf16* __restrict__ whbf,
    const __bf16* __restrict__ wwbf, const float* __restrict__ b_h,
    const float* __restrict__ h_work0, float* __restrict__ hwall,
    float* __restrict__ tape, float* __restrict__ hwf) {
  const int tid = threadIdx.x;
  const int wave = tid >> 5, lane = tid & 31;
  const int hf = lane >> 4, nloc = lane & 15;
  const int b0 = blockIdx.x * 16;

  extern __shared__ __align__(32) char smem[];
  __bf16* s_hw    = (__bf16*)(smem);
  float*  s_aux   = (float*)(smem + 16384);
  float*  s_wx    = (float*)(smem + 49152);
  float*  s_logit = (float*)(smem + 81920);
  float*  s_attn  = (float*)(smem + 86016);

  for (int idx = tid; idx < 16 * Dsz; idx += 256) {
    const int m = idx >> 9, d = idx & (Dsz - 1);
    s_hw[idx] = (__bf16)h_work0[(size_t)(b0 + m) * Dsz + d];
  }
  __syncthreads();

  const v8f z = {0.f, 0.f, 0.f, 0.f, 0.f, 0.f, 0.f, 0.f};

  for (int t = 0; t < Tsz; ++t) {
    // ---- kick off async LDS stage of Wx[:, t] (16x512 f32 = 2048 x b128);
    //      drains under the attention + read + WMMA K-loop below ----
    for (int q = tid; q < 2048; q += 256) {
      const int m = q >> 7;        // 128 x 16B chunks per row
      const int c = q & 127;
      const float* g = wx_all + ((size_t)(b0 + m) * Tsz + t) * Dsz + c * 4;
      async_g2lds_b128(s_wx + (size_t)q * 4, g);
    }

    // ---- read attention (old h_work, old tape) ----
    attn_pass(tid, b0, s_hw, tape, s_logit, s_attn);

    // ---- read[m,d] = sum_n attn[m,n] * tape[b0+m,n,d] -> s_aux ----
    for (int idx = tid; idx < 16 * Dsz; idx += 256) {
      const int m = idx >> 9, d = idx & (Dsz - 1);
      const float* tp = tape + ((size_t)(b0 + m) * Nsz) * Dsz + d;
      float s = 0.f;
#pragma unroll 4
      for (int n = 0; n < Nsz; ++n) s += s_attn[m * Nsz + n] * tp[(size_t)n * Dsz];
      s_aux[idx] = s;
    }
    __syncthreads();

    // ---- GEMM1: Rh = h_work @ W_h.T (WMMA) ----
    v8f acc[4] = {z, z, z, z};
    gemm16x512(wave, lane, s_hw, whbf, acc);
    wait_async0();     // this wave's Wx stage done
    __syncthreads();   // all waves: async data visible, old s_hw reads done
    // epilogue: h = tanh(Rh + Wx[:,t] + read + b_h)
#pragma unroll
    for (int nt = 0; nt < 4; ++nt) {
      const int n = (wave * 4 + nt) * 16 + nloc;
#pragma unroll
      for (int i = 0; i < 8; ++i) {
        const int m = i + 8 * hf;
        const size_t row = (size_t)(b0 + m) * Tsz + t;
        float v = acc[nt][i] + s_wx[(size_t)m * Dsz + n] + s_aux[(size_t)m * Dsz + n] + b_h[n];
        v = tanhf(v);
        hwall[row * Dsz + n] = v;
        if (t == Tsz - 1) hwf[(size_t)(b0 + m) * Dsz + n] = v;
        s_hw[(size_t)m * Dsz + n] = (__bf16)v;
      }
    }
    __syncthreads();

    // ---- GEMM2: write_val = new h_work @ W_write.T -> s_aux ----
    acc[0] = z; acc[1] = z; acc[2] = z; acc[3] = z;
    gemm16x512(wave, lane, s_hw, wwbf, acc);
    __syncthreads();
#pragma unroll
    for (int nt = 0; nt < 4; ++nt) {
      const int n = (wave * 4 + nt) * 16 + nloc;
#pragma unroll
      for (int i = 0; i < 8; ++i) {
        const int m = i + 8 * hf;
        s_aux[(size_t)m * Dsz + n] = acc[nt][i];
      }
    }
    __syncthreads();

    // ---- write attention (new h_work, old tape) ----
    attn_pass(tid, b0, s_hw, tape, s_logit, s_attn);

    // ---- tape = (1-wa)*tape + wa*write_val ----
    for (size_t idx = tid; idx < (size_t)16 * Nsz * Dsz; idx += 256) {
      const int d = (int)(idx & (Dsz - 1));
      const int n = (int)((idx >> 9) & (Nsz - 1));
      const int m = (int)(idx >> 15);
      const float a = s_attn[m * Nsz + n];
      const size_t off = ((size_t)(b0 + m) * Nsz + n) * Dsz + d;
      tape[off] = (1.f - a) * tape[off] + a * s_aux[(size_t)m * Dsz + d];
    }
    __syncthreads();
  }
}

// ---------------------------------------------------------------------------
// Host launcher
// ---------------------------------------------------------------------------
extern "C" void kernel_launch(void* const* d_in, const int* in_sizes, int n_in,
                              void* d_out, int out_size, void* d_ws, size_t ws_size,
                              hipStream_t stream) {
  (void)in_sizes; (void)n_in; (void)out_size; (void)ws_size;
  const float* x_seq  = (const float*)d_in[0];  // [B,T,D]
  const float* h_tape = (const float*)d_in[1];  // [B,N,D]
  const float* h_work = (const float*)d_in[2];  // [B,D]
  const float* W_x    = (const float*)d_in[3];  // [D,D]
  const float* W_h    = (const float*)d_in[4];  // [D,D]
  const float* W_w    = (const float*)d_in[5];  // [D,D]
  const float* b_h    = (const float*)d_in[6];  // [D]

  // Workspace layout
  char* ws = (char*)d_ws;
  float*  wx_all = (float*)ws;                                  // 64 MB
  __bf16* wxbf   = (__bf16*)(ws + (size_t)67108864);            // 512 KB
  __bf16* whbf   = (__bf16*)(ws + (size_t)67633152);            // 512 KB
  __bf16* wwbf   = (__bf16*)(ws + (size_t)68157440);            // 512 KB

  // Output layout: [h_work_all | h_tape_f | h_work_f]
  float* hwall = (float*)d_out;
  float* tape  = hwall + (size_t)Bsz * Tsz * Dsz;
  float* hwf   = tape + (size_t)Bsz * Nsz * Dsz;

  f2bf_kernel<<<256, 256, 0, stream>>>(W_x, wxbf, Dsz * Dsz);
  f2bf_kernel<<<256, 256, 0, stream>>>(W_h, whbf, Dsz * Dsz);
  f2bf_kernel<<<256, 256, 0, stream>>>(W_w, wwbf, Dsz * Dsz);
  // tape state evolves in-place inside the h_tape_f output region
  copyf_kernel<<<1024, 256, 0, stream>>>(h_tape, tape, (long)Bsz * Nsz * Dsz);

  dim3 g(Bsz * Tsz / 128, Dsz / 64, 1);  // 256 x 8
  wx_gemm_kernel<<<g, 256, 0, stream>>>(x_seq, wxbf, wx_all);

  recurrent_kernel<<<2, 256, RECUR_LDS_BYTES, stream>>>(
      wx_all, whbf, wwbf, b_h, h_work, hwall, tape, hwf);
}
